// SelfAdaptiveVoxelization_D_48284022342028
// MI455X (gfx1250) — matrix operand order repaired and checked by reading, not compiled, not run on hardware
//
#include <hip/hip_runtime.h>
#include <hip/hip_bf16.h>

typedef __attribute__((ext_vector_type(16))) _Float16 v16h;
typedef __attribute__((ext_vector_type(8)))  float    v8f;

#define CZv    1
#define CYv    400
#define CXv    352
#define NPTS   800000
#define CANVAS (4 * CZv * CYv * CXv)   // 563200
#define WAVES  8                       // waves per block (wave32)

// ---- workspace byte offsets -------------------------------------------------
// [w0 frags: 4 frag * 32 lane * 16 half]  4096 B
// [w1 frags: 16 frag * 32 lane * 16 half] 16384 B
// [sc0|sh0|sc1|sh1: 4*64 float]           1024 B
// [sumxyz: CANVAS*3 f32][v0: CANVAS*64 f32]
#define OFF_W0F   0
#define OFF_W1F   4096
#define OFF_SCSH  (4096 + 16384)
#define OFF_SUM   (OFF_SCSH + 1024)                    // 21504, 64B aligned
#define OFF_V0    (OFF_SUM + (size_t)CANVAS * 3 * 4)

// ---------------------------------------------------------------------------
// helpers
// ---------------------------------------------------------------------------
__device__ __forceinline__ int lin_idx(int4 c) {
    // coors row = (b, z, y, x)
    return ((c.x * CZv + c.y) * CYv + c.z) * CXv + c.w;
}

// 11-dim point feature (features, f_cluster, f_center, dist)
__device__ __forceinline__ void point_feats(int p,
        const float4* __restrict__ feats, const int4* __restrict__ coors,
        const float* __restrict__ counts, const float* __restrict__ sumxyz,
        float out[11], int* vox) {
    float4 f = feats[p];
    int4   c = coors[p];
    int  idx = lin_idx(c);
    *vox = idx;
    float safe = fmaxf(counts[idx], 1.0f);
    float inv  = 1.0f / safe;
    float mx = sumxyz[idx * 3 + 0] * inv;
    float my = sumxyz[idx * 3 + 1] * inv;
    float mz = sumxyz[idx * 3 + 2] * inv;
    out[0] = f.x; out[1] = f.y; out[2] = f.z; out[3] = f.w;
    out[4] = f.x - mx; out[5] = f.y - my; out[6] = f.z - mz;
    out[7] = f.x - ((float)c.w * 0.2f + 0.1f);      // VX, X_OFF
    out[8] = f.y - ((float)c.z * 0.2f - 39.9f);     // VY, Y_OFF
    out[9] = f.z - ((float)c.y * 4.0f - 1.0f);      // VZ, Z_OFF
    out[10] = sqrtf(f.x * f.x + f.y * f.y + f.z * f.z);
}

// write 11 feats (padded to 32 halfs) as 4x b128 into LDS row
__device__ __forceinline__ void store_feat_row(_Float16* __restrict__ row,
                                               const float fo[11]) {
    union { _Float16 h[32]; uint4 q[4]; } u;
#pragma unroll
    for (int k = 0; k < 11; ++k) u.h[k] = (_Float16)fo[k];
#pragma unroll
    for (int k = 11; k < 32; ++k) u.h[k] = (_Float16)0.0f;
    uint4* dst = (uint4*)row;
#pragma unroll
    for (int i = 0; i < 4; ++i) dst[i] = u.q[i];
}

// A fragment (16-bit 16xK row-major tile in LDS), ISA 16x32 A layout:
//   lane 0-15 : M=lane,    e=0..7 -> K=0..7,  e=8..15 -> K=16..23
//   lane16-31 : M=lane-16, e=0..7 -> K=8..15, e=8..15 -> K=24..31
__device__ __forceinline__ v16h load_afrag(const _Float16* __restrict__ lds,
                                           int ldk, int kbase, int lane) {
    union { v16h v; _Float16 h[16]; } u;
    int m   = lane & 15;
    int khi = (lane >> 4) << 3;
#pragma unroll
    for (int e = 0; e < 16; ++e) {
        int k = kbase + ((e < 8) ? e : (e + 8)) + khi;
        u.h[e] = lds[m * ldk + k];
    }
    return u.v;
}

// pre-swizzled B fragment (prep kernel wrote per-lane layout): 32B load
__device__ __forceinline__ v16h load_bfrag_pre(const _Float16* __restrict__ frags,
                                               int fragId, int lane) {
    return *(const v16h*)(frags + ((size_t)fragId * 32 + lane) * 16);
}

// ---------------------------------------------------------------------------
// kernel 0: pre-swizzle W0/W1 into WMMA B-fragment layout, fold BN params
// B layout: lane 0-15 : N=lane,    e=0..7 -> K=kbase+e,   e=8..15 -> K=kbase+16+(e-8)
//           lane16-31 : N=lane-16, e=0..7 -> K=kbase+8+e, e=8..15 -> K=kbase+24+(e-8)
// ---------------------------------------------------------------------------
__global__ __launch_bounds__(256) void k_prep(
        const float* __restrict__ W0, const float* __restrict__ g0,
        const float* __restrict__ b0, const float* __restrict__ rm0,
        const float* __restrict__ rv0,
        const float* __restrict__ W1, const float* __restrict__ g1,
        const float* __restrict__ b1, const float* __restrict__ rm1,
        const float* __restrict__ rv1,
        _Float16* __restrict__ w0f, _Float16* __restrict__ w1f,
        float* __restrict__ scsh) {
    // W0: 4 frags (nt), Ktot=11 padded to 32 -> 2048 halfs
    for (int t = threadIdx.x; t < 4 * 32 * 16; t += blockDim.x) {
        int e = t & 15, lane = (t >> 4) & 31, nt = t >> 9;
        int n = (lane & 15) + nt * 16;
        int k = ((lane >> 4) << 3) + ((e < 8) ? e : (e + 8));
        w0f[t] = (k < 11) ? (_Float16)W0[k * 64 + n] : (_Float16)0.0f;
    }
    // W1: 16 frags (kc*4+nt), Ktot=128 -> 8192 halfs
    for (int t = threadIdx.x; t < 16 * 32 * 16; t += blockDim.x) {
        int e = t & 15, lane = (t >> 4) & 31, f = t >> 9;
        int kc = f >> 2, nt = f & 3;
        int n = (lane & 15) + nt * 16;
        int k = kc * 32 + ((lane >> 4) << 3) + ((e < 8) ? e : (e + 8));
        w1f[t] = (_Float16)W1[k * 64 + n];
    }
    // folded BN affine: sc = g*rsqrt(rv+eps), sh = b - rm*sc
    if (threadIdx.x < 64) {
        int n = threadIdx.x;
        float s0 = g0[n] * rsqrtf(rv0[n] + 1e-3f);
        float s1 = g1[n] * rsqrtf(rv1[n] + 1e-3f);
        scsh[n]       = s0;
        scsh[64 + n]  = b0[n] - rm0[n] * s0;
        scsh[128 + n] = s1;
        scsh[192 + n] = b1[n] - rm1[n] * s1;
    }
}

// ---------------------------------------------------------------------------
// kernel 1: per-voxel count + xyz sums (float atomics, L2-resident)
// ---------------------------------------------------------------------------
__global__ __launch_bounds__(256) void k_scatter(
        const float4* __restrict__ feats, const int4* __restrict__ coors,
        float* __restrict__ counts, float* __restrict__ sumxyz) {
    int i = blockIdx.x * blockDim.x + threadIdx.x;
    if (i >= NPTS) return;
    int4 c = coors[i];
    int idx = lin_idx(c);
    float4 f = feats[i];
    atomicAdd(&counts[idx], 1.0f);
    atomicAdd(&sumxyz[idx * 3 + 0], f.x);
    atomicAdd(&sumxyz[idx * 3 + 1], f.y);
    atomicAdd(&sumxyz[idx * 3 + 2], f.z);
}

// ---------------------------------------------------------------------------
// kernel 2: VFE layer 0 via WMMA, scatter-max into v0 canvas
// ---------------------------------------------------------------------------
__global__ __launch_bounds__(256) void k_vfe0(
        const float4* __restrict__ feats, const int4* __restrict__ coors,
        const float* __restrict__ counts, const float* __restrict__ sumxyz,
        const _Float16* __restrict__ w0f, const float* __restrict__ scsh,
        float* __restrict__ v0) {
    __shared__ _Float16 ldsA[WAVES * 16 * 32];
    __shared__ int      sVox[WAVES * 16];
    int lane = threadIdx.x & 31;
    int wave = threadIdx.x >> 5;
    int pt0  = (blockIdx.x * WAVES + wave) * 16;
    _Float16* A = &ldsA[wave * 16 * 32];

    if (lane < 16) {                         // one point per low lane
        float fo[11]; int vox;
        point_feats(pt0 + lane, feats, coors, counts, sumxyz, fo, &vox);
        sVox[wave * 16 + lane] = vox;
        store_feat_row(&A[lane * 32], fo);
    }
    __syncthreads();

    v16h a   = load_afrag(A, 32, 0, lane);
    int mofs = (lane >> 4) << 3;             // C/D row offset per half-wave
    int n0   = lane & 15;
    size_t rowb[8];                          // hoisted voxel row bases
#pragma unroll
    for (int r = 0; r < 8; ++r)
        rowb[r] = (size_t)sVox[wave * 16 + mofs + r] * 64;
#pragma unroll
    for (int nt = 0; nt < 4; ++nt) {
        v16h b = load_bfrag_pre(w0f, nt, lane);
        v8f  c = {};
        c = __builtin_amdgcn_wmma_f32_16x16x32_f16(false, a, false, b,
                                                   (short)0, c, false, false);
        int   n  = n0 + nt * 16;
        float sc = scsh[n];
        float sh = scsh[64 + n];
#pragma unroll
        for (int r = 0; r < 8; ++r) {
            float val = fmaxf(fmaf(c[r], sc, sh), 0.0f);   // BN + ReLU >= 0
            atomicMax((unsigned int*)&v0[rowb[r] + n], __float_as_uint(val));
        }
    }
}

// ---------------------------------------------------------------------------
// kernel 3: recompute layer 0, concat with gathered v0, layer 1 via WMMA,
//           scatter-max into output canvas v1
// ---------------------------------------------------------------------------
__global__ __launch_bounds__(256) void k_vfe1(
        const float4* __restrict__ feats, const int4* __restrict__ coors,
        const float* __restrict__ counts, const float* __restrict__ sumxyz,
        const _Float16* __restrict__ w0f, const _Float16* __restrict__ w1f,
        const float* __restrict__ scsh,
        const float* __restrict__ v0, float* __restrict__ v1out) {
    __shared__ _Float16 ldsA0[WAVES * 16 * 32];
    __shared__ _Float16 ldsA1[WAVES * 16 * 128];
    __shared__ int      sVox[WAVES * 16];
    int lane = threadIdx.x & 31;
    int wave = threadIdx.x >> 5;
    int pt0  = (blockIdx.x * WAVES + wave) * 16;
    _Float16* A0 = &ldsA0[wave * 16 * 32];
    _Float16* A1 = &ldsA1[wave * 16 * 128];

    // phase 1: 11-dim features -> LDS (A-major)
    if (lane < 16) {
        float fo[11]; int vox;
        point_feats(pt0 + lane, feats, coors, counts, sumxyz, fo, &vox);
        sVox[wave * 16 + lane] = vox;
        store_feat_row(&A0[lane * 32], fo);
    }
    __syncthreads();

    // phase 2: layer 0 matmul -> pf0 into A1[:, 0:64]; gather v0 -> A1[:, 64:128]
    {
        v16h a   = load_afrag(A0, 32, 0, lane);
        int mofs = (lane >> 4) << 3;
        int n0   = lane & 15;
#pragma unroll
        for (int nt = 0; nt < 4; ++nt) {
            v16h b = load_bfrag_pre(w0f, nt, lane);
            v8f  c = {};
            c = __builtin_amdgcn_wmma_f32_16x16x32_f16(false, a, false, b,
                                                       (short)0, c, false, false);
            int   n  = n0 + nt * 16;
            float sc = scsh[n];
            float sh = scsh[64 + n];
#pragma unroll
            for (int r = 0; r < 8; ++r) {
                float val = fmaxf(fmaf(c[r], sc, sh), 0.0f);
                A1[(mofs + r) * 128 + n] = (_Float16)val;
            }
        }
        // cooperative gather of v0[idx] (L2-resident, 144 MB < 192 MB L2)
        int gm  = lane & 15;
        int gh  = lane >> 4;                 // half: cols 0..31 / 32..63
        int vox = sVox[wave * 16 + gm];
        const float4* src4 = (const float4*)&v0[(size_t)vox * 64 + gh * 32];
        unsigned long long* dst =
            (unsigned long long*)&A1[gm * 128 + 64 + gh * 32];
#pragma unroll
        for (int j = 0; j < 8; ++j) {
            float4 f = src4[j];
            union { _Float16 h[4]; unsigned long long u; } p;
            p.h[0] = (_Float16)f.x; p.h[1] = (_Float16)f.y;
            p.h[2] = (_Float16)f.z; p.h[3] = (_Float16)f.w;
            dst[j] = p.u;
        }
    }
    __syncthreads();

    // phase 3: layer 1 (K = 128 = 4 k-chunks) -> BN+ReLU -> atomicMax into v1
    v16h a1[4];
#pragma unroll
    for (int kc = 0; kc < 4; ++kc) a1[kc] = load_afrag(A1, 128, kc * 32, lane);
    int mofs = (lane >> 4) << 3;
    int n0   = lane & 15;
    size_t rowb[8];
#pragma unroll
    for (int r = 0; r < 8; ++r)
        rowb[r] = (size_t)sVox[wave * 16 + mofs + r] * 64;
#pragma unroll
    for (int nt = 0; nt < 4; ++nt) {
        v8f acc = {};
#pragma unroll
        for (int kc = 0; kc < 4; ++kc) {
            v16h b = load_bfrag_pre(w1f, kc * 4 + nt, lane);
            acc = __builtin_amdgcn_wmma_f32_16x16x32_f16(false, a1[kc], false, b,
                                                         (short)0, acc, false, false);
        }
        int   n  = n0 + nt * 16;
        float sc = scsh[128 + n];
        float sh = scsh[192 + n];
#pragma unroll
        for (int r = 0; r < 8; ++r) {
            float val = fmaxf(fmaf(acc[r], sc, sh), 0.0f);
            atomicMax((unsigned int*)&v1out[rowb[r] + n], __float_as_uint(val));
        }
    }
}

// ---------------------------------------------------------------------------
// launcher (graph-capture safe: memset nodes + kernels on `stream` only)
// ---------------------------------------------------------------------------
extern "C" void kernel_launch(void* const* d_in, const int* in_sizes, int n_in,
                              void* d_out, int out_size, void* d_ws, size_t ws_size,
                              hipStream_t stream) {
    const float4* feats = (const float4*)d_in[0];
    const int4*   coors = (const int4*)d_in[1];
    const float*  W0  = (const float*)d_in[2];
    const float*  g0  = (const float*)d_in[3];
    const float*  b0  = (const float*)d_in[4];
    const float*  rm0 = (const float*)d_in[5];
    const float*  rv0 = (const float*)d_in[6];
    const float*  W1  = (const float*)d_in[7];
    const float*  g1  = (const float*)d_in[8];
    const float*  b1  = (const float*)d_in[9];
    const float*  rm1 = (const float*)d_in[10];
    const float*  rv1 = (const float*)d_in[11];

    char* ws = (char*)d_ws;
    _Float16* w0f  = (_Float16*)(ws + OFF_W0F);
    _Float16* w1f  = (_Float16*)(ws + OFF_W1F);
    float*    scsh = (float*)   (ws + OFF_SCSH);
    float*    sumxyz = (float*) (ws + OFF_SUM);
    float*    v0     = (float*) (ws + OFF_V0);

    float* v1     = (float*)d_out;                       // [CANVAS, 64]
    float* counts = v1 + (size_t)CANVAS * 64;            // [CANVAS]

    // zero output canvas + counts and the scatter/v0 canvases, every call
    hipMemsetAsync(d_out, 0, (size_t)CANVAS * 65 * sizeof(float), stream);
    hipMemsetAsync(ws + OFF_SUM, 0, (size_t)CANVAS * 67 * sizeof(float), stream);

    k_prep<<<1, 256, 0, stream>>>(W0, g0, b0, rm0, rv0,
                                  W1, g1, b1, rm1, rv1, w0f, w1f, scsh);

    k_scatter<<<NPTS / 256, 256, 0, stream>>>(feats, coors, counts, sumxyz);

    int blocks = NPTS / (16 * WAVES);                    // 6250, exact
    k_vfe0<<<blocks, 256, 0, stream>>>(feats, coors, counts, sumxyz,
                                       w0f, scsh, v0);
    k_vfe1<<<blocks, 256, 0, stream>>>(feats, coors, counts, sumxyz,
                                       w0f, w1f, scsh, v0, v1);
}
